// LlamaAttentionWeight_69380901700300
// MI455X (gfx1250) — matrix operand order repaired and checked by reading, not compiled, not run on hardware
//
#include <hip/hip_runtime.h>

// ---------------------------------------------------------------------------
// Fused QKV projection (bf16 WMMA, f32 accumulate) + RoPE + head transpose
// for gfx1250 (CDNA5, wave32).
// Workgroup tile 256(M) x 128(N); each wave owns 32 rows x full head (128 cols)
// => LDS bytes per WMMA drops from 1.125KB to 0.625KB (LDS was the bottleneck).
// Double-buffered LDS; B fragments processed in 2 groups of 4 to cap VGPRs.
// ---------------------------------------------------------------------------

typedef __attribute__((ext_vector_type(16))) __bf16 v16bf;
typedef __attribute__((ext_vector_type(8)))  __bf16 v8bf;
typedef __attribute__((ext_vector_type(8)))  float  v8f;

#define D_MODEL   2048
#define T_SEQ     8192
#define BATCH     2
#define N_HEADS   16
#define N_KVHEADS 4
#define HEAD_DIM  128
#define NCOLS     3072           // 2048 q + 512 k + 512 v
#define BM        256
#define BN        128            // == HEAD_DIM, head-aligned
#define BK        32
#define NKITER    (D_MODEL / BK) // 64
#define LDSROW    40             // bf16 elems per row (32 + 8 pad; 80B rows, 16B-aligned chunks)

// ln(10000)/64
#define ROPE_LOG_STEP 0.14391156831212787f

struct TileX { float4 v[8]; };   // 32 floats: one full BK row of x per thread
struct TileW { float4 v[4]; };   // 16 floats: half a BK row of W per thread

__device__ __forceinline__ TileX ldx(const float* p) {
    TileX t;
#pragma unroll
    for (int i = 0; i < 8; ++i) t.v[i] = ((const float4*)p)[i];
    return t;
}
__device__ __forceinline__ TileW ldw(const float* p) {
    TileW t;
#pragma unroll
    for (int i = 0; i < 4; ++i) t.v[i] = ((const float4*)p)[i];
    return t;
}

__device__ __forceinline__ v8bf cvt8(float4 a, float4 b) {
    v8bf r;
    r[0] = (__bf16)a.x; r[1] = (__bf16)a.y; r[2] = (__bf16)a.z; r[3] = (__bf16)a.w;
    r[4] = (__bf16)b.x; r[5] = (__bf16)b.y; r[6] = (__bf16)b.z; r[7] = (__bf16)b.w;
    return r;
}

__device__ __forceinline__ void stx(__bf16* d, const TileX& t) {
#pragma unroll
    for (int i = 0; i < 4; ++i)
        *(v8bf*)(d + 8 * i) = cvt8(t.v[2 * i], t.v[2 * i + 1]);
}
__device__ __forceinline__ void stw(__bf16* d, const TileW& t) {
    *(v8bf*)(d)     = cvt8(t.v[0], t.v[1]);
    *(v8bf*)(d + 8) = cvt8(t.v[2], t.v[3]);
}

// A fragment: 16x32 bf16, M = row; K = 16*hi + j (contiguous 16)
__device__ __forceinline__ v16bf loadA(const __bf16* xsBuf, int row, int hi) {
    const __bf16* ar = xsBuf + row * LDSROW;
    v8bf a0 = *(const v8bf*)(ar + 16 * hi);
    v8bf a1 = *(const v8bf*)(ar + 16 * hi + 8);
    return __builtin_shufflevector(a0, a1,
               0, 1, 2, 3, 4, 5, 6, 7, 8, 9, 10, 11, 12, 13, 14, 15);
}

// B fragment: 32x16 bf16, N = nt*16 + lo, K = 16*hi + j (contiguous 16)
__device__ __forceinline__ v16bf loadB(const __bf16* wsBuf, int nt, int lo, int hi) {
    const __bf16* br = wsBuf + (nt * 16 + lo) * LDSROW + 16 * hi;
    v8bf b0 = *(const v8bf*)(br);
    v8bf b1 = *(const v8bf*)(br + 8);
    return __builtin_shufflevector(b0, b1,
               0, 1, 2, 3, 4, 5, 6, 7, 8, 9, 10, 11, 12, 13, 14, 15);
}

__device__ __forceinline__ void mma_block(const __bf16* curX, const __bf16* curW,
                                          v8f (&acc)[2][8], int arow0, int lo, int hi)
{
    v16bf a0 = loadA(curX, arow0, hi);
    v16bf a1 = loadA(curX, arow0 + 16, hi);
#pragma unroll
    for (int g = 0; g < 2; ++g) {
        v16bf bf[4];
#pragma unroll
        for (int j = 0; j < 4; ++j) bf[j] = loadB(curW, g * 4 + j, lo, hi);
#pragma unroll
        for (int j = 0; j < 4; ++j) {
            acc[0][g * 4 + j] = __builtin_amdgcn_wmma_f32_16x16x32_bf16(
                false, a0, false, bf[j], (short)0, acc[0][g * 4 + j], false, false);
            acc[1][g * 4 + j] = __builtin_amdgcn_wmma_f32_16x16x32_bf16(
                false, a1, false, bf[j], (short)0, acc[1][g * 4 + j], false, false);
        }
    }
}

// One K-step: global-load next tile, WMMA on current buffer, convert+store next
// tile into the other buffer, barrier.
__device__ __forceinline__ void kstep(const float*& xrow, const float*& wrow,
                                      const __bf16* curX, const __bf16* curW,
                                      __bf16* nxtXdst, __bf16* nxtWdst,
                                      v8f (&acc)[2][8], int arow0, int lo, int hi)
{
    TileX gx = ldx(xrow);
    TileW gw = ldw(wrow);
    xrow += BK; wrow += BK;

    mma_block(curX, curW, acc, arow0, lo, hi);

    stx(nxtXdst, gx);
    stw(nxtWdst, gw);
    __syncthreads();
}

__global__ __launch_bounds__(256)
void qkv_rope_wmma_kernel(const float* __restrict__ x,
                          const float* __restrict__ wq,
                          const float* __restrict__ wk,
                          const float* __restrict__ wv,
                          const int*   __restrict__ pos_offset_p,
                          float*       __restrict__ out)
{
    __shared__ __bf16 lds[2 * (BM + BN) * LDSROW];  // 60 KB double-buffered tiles
    __bf16* xs0 = lds;
    __bf16* ws0 = lds + BM * LDSROW;
    __bf16* xs1 = lds + (BM + BN) * LDSROW;
    __bf16* ws1 = lds + (2 * BM + BN) * LDSROW;

    const int tileM = blockIdx.x;        // 0..63
    const int tileN = blockIdx.y;        // 0..23 (one head per tile)
    const int tid   = threadIdx.x;       // 0..255
    const int wave  = tid >> 5;          // 0..7 -> rows [wave*32, wave*32+32)
    const int lane  = tid & 31;
    const int lo    = lane & 15;
    const int hi    = lane >> 4;
    const int arow0 = wave * 32 + lo;    // first A-fragment row for this wave/lane

    // ---- weight region for this N tile (128-aligned => single region)
    const float* wbase;
    int region, head;                    // 0 = q (rope), 1 = k (rope), 2 = v
    if (tileN < 16)      { wbase = wq + (size_t)tileN        * HEAD_DIM * D_MODEL; region = 0; head = tileN; }
    else if (tileN < 20) { wbase = wk + (size_t)(tileN - 16) * HEAD_DIM * D_MODEL; region = 1; head = tileN - 16; }
    else                 { wbase = wv + (size_t)(tileN - 20) * HEAD_DIM * D_MODEL; region = 2; head = tileN - 20; }

    // ---- staging: x = one full 32-col row per thread; W = (row, 16-col half)
    const float* xrow = x + ((size_t)(tileM * BM + tid)) * D_MODEL;
    const int wsrow = tid >> 1;
    const int wscol = (tid & 1) * 16;
    const float* wrow = wbase + ((size_t)wsrow) * D_MODEL + wscol;

    __bf16* xd0 = xs0 + tid * LDSROW;
    __bf16* xd1 = xs1 + tid * LDSROW;
    __bf16* wd0 = ws0 + wsrow * LDSROW + wscol;
    __bf16* wd1 = ws1 + wsrow * LDSROW + wscol;

    v8f acc[2][8] = {};

    // ---- prologue: tile 0 -> buf0
    {
        TileX gx = ldx(xrow);
        TileW gw = ldw(wrow);
        xrow += BK; wrow += BK;
        stx(xd0, gx);
        stw(wd0, gw);
        __syncthreads();
    }

    // ---- steady state: 31 ping-pong pairs (tiles 0..61)
    for (int i = 0; i < (NKITER - 2) / 2; ++i) {
        kstep(xrow, wrow, xs0, ws0, xd1, wd1, acc, arow0, lo, hi);  // read buf0, fill buf1
        kstep(xrow, wrow, xs1, ws1, xd0, wd0, acc, arow0, lo, hi);  // read buf1, fill buf0
    }
    // tile 62 from buf0, load tile 63 -> buf1
    kstep(xrow, wrow, xs0, ws0, xd1, wd1, acc, arow0, lo, hi);

    // ---- final tile (63) from buf1, no further loads
    mma_block(xs1, ws1, acc, arow0, lo, hi);

    // ------------------------------------------------------------------ epilogue
    const int pos_off = *pos_offset_p;

    const int rowBase = tileM * BM + wave * 32;       // 32-row strip; 8192 % 256 == 0
    const int b       = rowBase / T_SEQ;
    const int tBase   = rowBase % T_SEQ;

    const size_t QSZ = (size_t)BATCH * N_HEADS   * T_SEQ * HEAD_DIM;
    const size_t KSZ = (size_t)BATCH * N_KVHEADS * T_SEQ * HEAD_DIM;
    size_t planeBase;
    if (region == 0)      planeBase = ((size_t)(b * N_HEADS   + head)) * T_SEQ * HEAD_DIM;
    else if (region == 1) planeBase = QSZ + ((size_t)(b * N_KVHEADS + head)) * T_SEQ * HEAD_DIM;
    else                  planeBase = QSZ + KSZ + ((size_t)(b * N_KVHEADS + head)) * T_SEQ * HEAD_DIM;

    if (region < 2) {
        // RoPE: col d = nt*16+lo (nt<4) pairs with d+64 in acc[m][nt+4], same lane/row.
#pragma unroll
        for (int nt = 0; nt < 4; ++nt) {
            const int   fi   = nt * 16 + lo;                       // freq index 0..63
            const float invf = __expf((float)fi * -ROPE_LOG_STEP);
#pragma unroll
            for (int m = 0; m < 2; ++m) {
#pragma unroll
                for (int r = 0; r < 8; ++r) {
                    const int t = tBase + m * 16 + r + 8 * hi;
                    float s, c;
                    __sincosf((float)(t + pos_off) * invf, &s, &c);
                    const float vlo = acc[m][nt][r];
                    const float vhi = acc[m][nt + 4][r];
                    const size_t o  = planeBase + (size_t)t * HEAD_DIM;
                    out[o + nt * 16 + lo]      = vlo * c - vhi * s;
                    out[o + nt * 16 + lo + 64] = vhi * c + vlo * s;
                }
            }
        }
    } else {
        // V: straight transpose-scatter (lane = d -> coalesced 64B rows)
#pragma unroll
        for (int nt = 0; nt < 8; ++nt) {
#pragma unroll
            for (int m = 0; m < 2; ++m) {
#pragma unroll
                for (int r = 0; r < 8; ++r) {
                    const int t = tBase + m * 16 + r + 8 * hi;
                    out[planeBase + (size_t)t * HEAD_DIM + nt * 16 + lo] = acc[m][nt][r];
                }
            }
        }
    }
}

extern "C" void kernel_launch(void* const* d_in, const int* in_sizes, int n_in,
                              void* d_out, int out_size, void* d_ws, size_t ws_size,
                              hipStream_t stream) {
    (void)in_sizes; (void)n_in; (void)out_size; (void)d_ws; (void)ws_size;
    const float* x   = (const float*)d_in[0];
    const float* wq  = (const float*)d_in[1];
    const float* wk  = (const float*)d_in[2];
    const float* wv  = (const float*)d_in[3];
    const int*   pos = (const int*)d_in[4];

    dim3 grid((BATCH * T_SEQ) / BM, NCOLS / BN);   // (64, 24)
    dim3 block(256);
    qkv_rope_wmma_kernel<<<grid, block, 0, stream>>>(x, wq, wk, wv, pos, (float*)d_out);
}
